// LatticeLSTM_27908697489786
// MI455X (gfx1250) — matrix operand order repaired and checked by reading, not compiled, not run on hardware
//
#include <hip/hip_runtime.h>

typedef __bf16 bf16_t;
typedef __attribute__((ext_vector_type(16))) __bf16 v16bf;
typedef __attribute__((ext_vector_type(8)))  float  v8f;

#define S_LEN 2048
#define D_DIM 512
#define H_DIM 512
#define H3    1536
#define KW_N  3
#define KMAX  24

// ---------------- f32 -> bf16 conversion (row-major copy) ----------------
__global__ void cvt_f32_bf16(const float* __restrict__ in, bf16_t* __restrict__ out, int n) {
    int i = blockIdx.x * blockDim.x + threadIdx.x;
    if (i < n) out[i] = (bf16_t)in[i];
}

// ---------------- f32 -> bf16 convert + transpose: in[K,N] -> out[N,K] ----------------
__global__ void cvt_transpose_bf16(const float* __restrict__ in, bf16_t* __restrict__ out,
                                   int K, int N) {
    int i = blockIdx.x * blockDim.x + threadIdx.x;
    if (i < K * N) {
        int k = i / N;
        int n = i - k * N;
        out[(size_t)n * K + k] = (bf16_t)in[i];
    }
}

// ---------------- embedding gather + convert ----------------
// out[m*512+k] = bf16(emb[ids[m]*512 + k]),  m in [0, S*KW)
__global__ void gather_emb_bf16(const float* __restrict__ emb, const int* __restrict__ ids,
                                bf16_t* __restrict__ out, int n) {
    int i = blockIdx.x * blockDim.x + threadIdx.x;
    if (i < n) {
        int m = i >> 9;
        int k = i & 511;
        out[i] = (bf16_t)emb[(size_t)ids[m] * 512 + k];
    }
}

// ---------------- bf16 WMMA GEMM with fused bias ----------------
// C[M,N] = A[M,K] @ B[K,N] + bias[N], B pre-transposed as Bt[N,K].
// Each wave computes a 64x16 C strip (4 M-tiles x 1 N-tile): one B fragment
// (2x global_load_b128, contiguous) feeds 4 v_wmma issues.
__global__ void __launch_bounds__(256)
wmma_gemm_bias(const bf16_t* __restrict__ A, const bf16_t* __restrict__ Bt,
               const float* __restrict__ bias, float* __restrict__ C,
               int M, int N, int K) {
    int wave    = (blockIdx.x * blockDim.x + threadIdx.x) >> 5;
    int lane    = threadIdx.x & 31;
    int tiles_n = N >> 4;
    int strip   = wave / tiles_n;          // 64-row M strip
    int tn      = wave % tiles_n;
    int half    = lane >> 4;               // 0: lanes 0-15, 1: lanes 16-31
    int lr      = lane & 15;

    int col = tn * 16 + lr;                               // B/C column for this lane
    const bf16_t* Bcol = Bt + (size_t)col * K;            // contiguous K for this column

    const bf16_t* Ar[4];
    #pragma unroll
    for (int m = 0; m < 4; ++m)
        Ar[m] = A + (size_t)(strip * 64 + m * 16 + lr) * K;

    v8f acc[4] = {};
    for (int kt = 0; kt < K; kt += 32) {
        // B 32x16 fragment: lanes 0-15 hold K=kt..kt+15, lanes 16-31 K=kt+16..kt+31
        v16bf bfr;
        const bf16_t* bp = Bcol + kt + half * 16;
        #pragma unroll
        for (int e = 0; e < 16; ++e) bfr[e] = bp[e];

        #pragma unroll
        for (int m = 0; m < 4; ++m) {
            // A 16x32 fragment: lane-half selects K {0..15} vs {16..31}
            v16bf af;
            const bf16_t* ap = Ar[m] + kt + half * 16;
            #pragma unroll
            for (int e = 0; e < 8; ++e) af[e]     = ap[e];
            #pragma unroll
            for (int e = 0; e < 8; ++e) af[8 + e] = ap[8 + e];
            acc[m] = __builtin_amdgcn_wmma_f32_16x16x32_bf16(
                         false, af, false, bfr, (short)0, acc[m], false, false);
        }
    }

    float bv = bias ? bias[col] : 0.0f;
    #pragma unroll
    for (int m = 0; m < 4; ++m) {
        #pragma unroll
        for (int r = 0; r < 8; ++r) {   // VGPR r -> rows r (lanes 0-15) / r+8 (lanes 16-31)
            int row = strip * 64 + m * 16 + half * 8 + r;
            C[(size_t)row * N + col] = acc[m][r] + bv;
        }
    }
}

// ---------------- sequential lattice scan (single persistent workgroup) ---------------
// LDS: h[512] | cpr[512] | ring[8*KW*512] | gates[1536] | hw[1536] | aw[KMAX*512]
//      | msk[KMAX] | base[KMAX] | anyw
__global__ void __launch_bounds__(1024)
lattice_scan(const float* __restrict__ WI,  const float* __restrict__ AWI,
             const float* __restrict__ WWI,
             const float* __restrict__ Whh, const float* __restrict__ aWhh,
             const float* __restrict__ Wwhh,
             const int* __restrict__ gp, const int* __restrict__ gs,
             const unsigned char* __restrict__ gm, int Km,
             float* __restrict__ hs, float* __restrict__ cs) {
    extern __shared__ float sm[];
    float* h     = sm;                    // 512
    float* cpr   = h + 512;               // 512
    float* ring  = cpr + 512;             // 8*KW*512 = 12288
    float* gates = ring + 8 * KW_N * 512; // 1536
    float* hw    = gates + H3;            // 1536
    float* aw    = hw + H3;               // KMAX*512
    float* msk   = aw + KMAX * 512;       // KMAX
    int*   base  = (int*)(msk + KMAX);    // KMAX
    int*   anyw  = base + KMAX;           // 1

    const int tid = threadIdx.x;
    for (int i = tid; i < 512 + 512 + 8 * KW_N * 512; i += 1024) sm[i] = 0.0f;
    __syncthreads();

    if (Km > KMAX) Km = KMAX;

    for (int t = 0; t < S_LEN; ++t) {
        // ---- setup: gather descriptors for words ending at t ----
        if (tid < Km) {
            int k = tid;
            base[k] = (((gp[t * Km + k]) & 7) * KW_N + gs[t * Km + k]) * 512;
            msk[k]  = gm[t * Km + k] ? 1.0f : 0.0f;
        }
        if (tid == 0) {
            int a = 0;
            for (int k = 0; k < Km; ++k) a |= (int)gm[t * Km + k];
            anyw[0] = a;
        }
        if (t + 1 < S_LEN) {  // warm L2/WGP$ for next step's activation rows
            __builtin_prefetch(&WI[(size_t)(t + 1) * H3 + (tid % H3)], 0, 1);
        }
        __syncthreads();

        // ---- char-cell gates GEMV: gates = WI[t] (+b folded) + h @ W_hh ----
        for (int n = tid; n < H3; n += 1024) {
            float acc = WI[(size_t)t * H3 + n];
            for (int k = 0; k < H_DIM; ++k) acc += h[k] * Whh[(size_t)k * H3 + n];
            gates[n] = acc;
        }
        // ---- alpha weights: aw[k][n] = exp(sigmoid(AWI[t] + c_in_k @ aW_hh)) * mask ----
        for (int idx = tid; idx < Km * 512; idx += 1024) {
            int k = idx >> 9, n = idx & 511;
            const float* cin = ring + base[k];
            float acc = AWI[(size_t)t * 512 + n];
            for (int kk = 0; kk < H_DIM; ++kk) acc += cin[kk] * aWhh[(size_t)kk * 512 + n];
            float al = 1.0f / (1.0f + expf(-acc));
            aw[idx] = expf(al) * msk[k];
        }
        __syncthreads();

        // ---- exp-normalized blend + state update ----
        if (tid < 512) {
            int n = tid;
            float ig = 1.0f / (1.0f + expf(-gates[n]));
            float og = 1.0f / (1.0f + expf(-gates[512 + n]));
            float gg = tanhf(gates[1024 + n]);
            float wi = expf(ig);
            float num = wi * gg, den = wi;
            for (int k = 0; k < Km; ++k) {
                float a = aw[k * 512 + n];
                num += a * ring[base[k] + n];
                den += a;
            }
            float cm  = num / den;
            float cpl = (1.0f - ig) * cpr[n] + ig * gg;
            float c1  = anyw[0] ? cm : cpl;
            float h1  = og * tanhf(c1);
            hs[(size_t)t * 512 + n] = h1;
            cs[(size_t)t * 512 + n] = c1;
            h[n]   = h1;
            cpr[n] = c1;
        }
        __syncthreads();

        // ---- word-cell hidden projection: hw = h1 @ Ww_hh (bw folded into WWI) ----
        for (int n = tid; n < H3; n += 1024) {
            float acc = 0.0f;
            for (int k = 0; k < H_DIM; ++k) acc += h[k] * Wwhh[(size_t)k * H3 + n];
            hw[n] = acc;
        }
        __syncthreads();

        // ---- word cells starting at t -> ring slot t&7 ----
        for (int idx = tid; idx < KW_N * 512; idx += 1024) {
            int j = idx >> 9, n = idx & 511;
            const float* wrow = WWI + ((size_t)t * KW_N + j) * H3;
            float f  = 1.0f / (1.0f + expf(-(wrow[n]        + hw[n])));
            float iw = 1.0f / (1.0f + expf(-(wrow[512 + n]  + hw[512 + n])));
            float gw = tanhf(wrow[1024 + n] + hw[1024 + n]);
            ring[((t & 7) * KW_N + j) * 512 + n] = f * cpr[n] + iw * gw;
        }
        __syncthreads();
    }
}

// ---------------------------------------------------------------------------
extern "C" void kernel_launch(void* const* d_in, const int* in_sizes, int n_in,
                              void* d_out, int out_size, void* d_ws, size_t ws_size,
                              hipStream_t stream) {
    const float* x        = (const float*)d_in[0];
    const float* W_ih     = (const float*)d_in[1];
    const float* W_hh     = (const float*)d_in[2];
    const float* b        = (const float*)d_in[3];
    const float* aW_ih    = (const float*)d_in[4];
    const float* aW_hh    = (const float*)d_in[5];
    const float* ab       = (const float*)d_in[6];
    const float* Ww_ih    = (const float*)d_in[7];
    const float* Ww_hh    = (const float*)d_in[8];
    const float* bw       = (const float*)d_in[9];
    const float* word_emb = (const float*)d_in[10];
    const int*   word_ids = (const int*)d_in[11];
    const int*   gpos     = (const int*)d_in[12];
    const int*   gslot    = (const int*)d_in[13];
    const unsigned char* gmask = (const unsigned char*)d_in[14];

    const int Km = in_sizes[12] / S_LEN;
    const int MW = S_LEN * KW_N;   // 6144 gathered word rows

    // workspace layout
    char* ws = (char*)d_ws;
    bf16_t* xbf    = (bf16_t*)ws;  ws += (size_t)S_LEN * D_DIM * sizeof(bf16_t);
    bf16_t* WihT   = (bf16_t*)ws;  ws += (size_t)H3    * D_DIM * sizeof(bf16_t); // [N,K]
    bf16_t* aWihT  = (bf16_t*)ws;  ws += (size_t)H_DIM * D_DIM * sizeof(bf16_t); // [N,K]
    bf16_t* WwihT  = (bf16_t*)ws;  ws += (size_t)H3    * D_DIM * sizeof(bf16_t); // [N,K]
    bf16_t* wembb  = (bf16_t*)ws;  ws += (size_t)MW    * D_DIM * sizeof(bf16_t);
    float*  WI     = (float*)ws;   ws += (size_t)S_LEN * H3    * sizeof(float);
    float*  AWI    = (float*)ws;   ws += (size_t)S_LEN * H_DIM * sizeof(float);
    float*  WWI    = (float*)ws;   ws += (size_t)MW    * H3    * sizeof(float);

    float* hs = (float*)d_out;
    float* cs = hs + (size_t)S_LEN * H_DIM;

    // conversions / transposes / gather (one-time, cheap)
    {
        int n;
        n = S_LEN * D_DIM; cvt_f32_bf16<<<(n + 255) / 256, 256, 0, stream>>>(x, xbf, n);
        n = D_DIM * H3;    cvt_transpose_bf16<<<(n + 255) / 256, 256, 0, stream>>>(W_ih,  WihT,  D_DIM, H3);
        n = D_DIM * H_DIM; cvt_transpose_bf16<<<(n + 255) / 256, 256, 0, stream>>>(aW_ih, aWihT, D_DIM, H_DIM);
        n = D_DIM * H3;    cvt_transpose_bf16<<<(n + 255) / 256, 256, 0, stream>>>(Ww_ih, WwihT, D_DIM, H3);
        n = MW * D_DIM;    gather_emb_bf16<<<(n + 255) / 256, 256, 0, stream>>>(word_emb, word_ids, wembb, n);
    }

    // WMMA pre-projections (bias folded); each wave = 64x16 strip
    {
        int waves, blocks;
        waves  = (S_LEN / 64) * (H3 / 16);     // 3072
        blocks = waves / 8;
        wmma_gemm_bias<<<blocks, 256, 0, stream>>>(xbf, WihT, b, WI, S_LEN, H3, D_DIM);
        waves  = (S_LEN / 64) * (H_DIM / 16);  // 1024
        blocks = waves / 8;
        wmma_gemm_bias<<<blocks, 256, 0, stream>>>(xbf, aWihT, ab, AWI, S_LEN, H_DIM, D_DIM);
        waves  = (MW / 64) * (H3 / 16);        // 9216
        blocks = waves / 8;
        wmma_gemm_bias<<<blocks, 256, 0, stream>>>(wembb, WwihT, bw, WWI, MW, H3, D_DIM);
    }

    // sequential scan: one persistent workgroup, big LDS (ring buffer depth 8)
    {
        size_t smem = (size_t)(512 + 512 + 8 * KW_N * 512 + H3 + H3 + KMAX * 512) * sizeof(float)
                      + (size_t)KMAX * sizeof(float)   // msk
                      + (size_t)KMAX * sizeof(int)     // base
                      + 16;                            // anyw + pad
        lattice_scan<<<1, 1024, smem, stream>>>(WI, AWI, WWI, W_hh, aW_hh, Ww_hh,
                                                gpos, gslot, gmask, Km, hs, cs);
    }
}
